// LEAM_66503273611447
// MI455X (gfx1250) — compile-verified, matrix operand order checked by reading, not run on hardware
//
#include <hip/hip_runtime.h>
#include <hip/hip_bf16.h>
#include <math.h>

// Problem constants (match reference)
#define Bn    64
#define Sn    512
#define Pn    300
#define Pp    320      // P padded to 10 chunks of 32 (WMMA K)
#define Kn    50
#define Kp    64       // K padded to 4 tiles of 16
#define Rr    5

typedef __attribute__((ext_vector_type(16))) _Float16 v16h;
typedef __attribute__((ext_vector_type(8)))  float    v8f;

union V16H { v16h v; uint4 q[2]; };

// ---------------------------------------------------------------------------
// Kernel 1/2: L2-normalize rows (either C directly, or gathered emb rows),
// write f16 rows padded to Pp=320 (zeros in 300..319; zero rows for k>=validRows)
// One wave (32 lanes) per row; 8 waves per block.
// ---------------------------------------------------------------------------
__global__ void leam_norm_rows(const float* __restrict__ base,
                               const int*   __restrict__ gidx,   // nullptr => row index
                               _Float16*    __restrict__ dst,    // [nrows][Pp]
                               int nrows, int validRows)
{
    const int row  = blockIdx.x * 8 + (threadIdx.x >> 5);
    const int lane = threadIdx.x & 31;
    if (row >= nrows) return;
    _Float16* d = dst + (size_t)row * Pp;

    if (row >= validRows) {                 // K-padding rows: all zero
        for (int p = lane; p < Pp; p += 32) d[p] = (_Float16)0.0f;
        return;
    }
    const int srcRow = gidx ? gidx[row] : row;
    const float* s = base + (size_t)srcRow * Pn;

    float v[10];
    float ss = 0.0f;
    #pragma unroll
    for (int i = 0; i < 10; ++i) {
        const int p = lane + i * 32;
        const float t = (p < Pn) ? s[p] : 0.0f;
        v[i] = t;
        ss += t * t;
    }
    #pragma unroll
    for (int off = 16; off > 0; off >>= 1) ss += __shfl_down(ss, off, 32);
    ss = __shfl(ss, 0, 32);
    const float inv = 1.0f / (sqrtf(ss) + 0.001f);
    #pragma unroll
    for (int i = 0; i < 10; ++i) {
        const int p = lane + i * 32;
        d[p] = (_Float16)((p < Pn) ? v[i] * inv : 0.0f);
    }
}

// ---------------------------------------------------------------------------
// Kernel 3: WMMA GEMM  G[b, k, s] = sum_p lhat[k,p] * xhat[b*S+s, p]
// Block = 256 threads (8 waves); block owns (b, 64-wide S tile).
// Wave (w) -> M-tile = w&3, N-tiles {2*(w>>2), 2*(w>>2)+1}  (4x4 16x16 tiles).
// A fragment (16-bit A 16x32 layout): lane<16 -> K {0..7,16..23}, lane>=16 ->
// K {8..15,24..31}; B fragment mirrors with N striped across lanes.
// ---------------------------------------------------------------------------
__global__ void leam_gemm_wmma(const _Float16* __restrict__ lhat, // [Kp][Pp]
                               const _Float16* __restrict__ xhat, // [B*S][Pp]
                               float*          __restrict__ G)    // [B][Kp][S]
{
    const int blk  = blockIdx.x;
    const int b    = blk >> 3;            // 8 S-tiles per batch
    const int s0   = (blk & 7) << 6;      // 64-wide S tile
    const int wave = threadIdx.x >> 5;
    const int lane = threadIdx.x & 31;
    const int m0   = (wave & 3) << 4;
    const int nb   = (wave >> 2) << 1;    // first of 2 N-tiles: 0 or 2
    const int lrow = lane & 15;
    const bool lo  = lane < 16;

    v8f acc0 = {};
    v8f acc1 = {};

    const _Float16* aRow  = lhat + (size_t)(m0 + lrow) * Pp;
    const _Float16* bRow0 = xhat + (size_t)(b * Sn + s0 + nb * 16 + lrow) * Pp;
    const _Float16* bRow1 = bRow0 + (size_t)16 * Pp;

    #pragma unroll 2
    for (int c = 0; c < Pp / 32; ++c) {
        const int p0  = c * 32;
        const int klo = p0 + (lo ? 0 : 8);
        const int khi = p0 + (lo ? 16 : 24);
        V16H a, f0, f1;
        a.q[0]  = *(const uint4*)(aRow  + klo);
        a.q[1]  = *(const uint4*)(aRow  + khi);
        f0.q[0] = *(const uint4*)(bRow0 + klo);
        f0.q[1] = *(const uint4*)(bRow0 + khi);
        f1.q[0] = *(const uint4*)(bRow1 + klo);
        f1.q[1] = *(const uint4*)(bRow1 + khi);
        acc0 = __builtin_amdgcn_wmma_f32_16x16x32_f16(
                 false, a.v, false, f0.v, (short)0, acc0, false, false);
        acc1 = __builtin_amdgcn_wmma_f32_16x16x32_f16(
                 false, a.v, false, f1.v, (short)0, acc1, false, false);
    }

    // C/D layout: VGPR r, lanes 0-15 -> M=r, lanes 16-31 -> M=8+r; N = lane%16
    float* Gb = G + (size_t)b * Kp * Sn;
    const int mBase = m0 + (lo ? 0 : 8);
    const int col0  = s0 + nb * 16 + lrow;
    #pragma unroll
    for (int r = 0; r < 8; ++r) {
        Gb[(size_t)(mBase + r) * Sn + col0]      = acc0[r];
        Gb[(size_t)(mBase + r) * Sn + col0 + 16] = acc1[r];
    }
}

// ---------------------------------------------------------------------------
// Kernel 4: width-11 cross-correlation along S (+bias), ReLU, max over k<50.
// One thread per (b,s); G tile is hot in L2/L0 (consecutive s share reads).
// ---------------------------------------------------------------------------
__global__ void leam_conv_max(const float* __restrict__ G,       // [B][Kp][S]
                              const float* __restrict__ conv_w,  // [11]
                              const float* __restrict__ conv_b,  // [1]
                              float*       __restrict__ mOut)    // [B][S]
{
    const int g = blockIdx.x * blockDim.x + threadIdx.x;
    const int b = g >> 9;
    const int s = g & (Sn - 1);

    float w[2 * Rr + 1];
    #pragma unroll
    for (int j = 0; j < 2 * Rr + 1; ++j) w[j] = conv_w[j];
    const float bias = conv_b[0];

    const float* Gb = G + (size_t)b * Kp * Sn;
    float best = 0.0f;                     // ReLU outputs are >= 0
    for (int k = 0; k < Kn; ++k) {
        const float* row = Gb + (size_t)k * Sn;
        float acc = bias;
        #pragma unroll
        for (int j = 0; j < 2 * Rr + 1; ++j) {
            const int ss = s - Rr + j;
            acc += (ss >= 0 && ss < Sn) ? w[j] * row[ss] : 0.0f;
        }
        best = fmaxf(best, fmaxf(acc, 0.0f));
    }
    mOut[g] = best;
}

// ---------------------------------------------------------------------------
// Kernel 5: per batch b: softmax over S, attention-weighted mean pool of the
// re-gathered f32 emb rows (coalesced over P), then out = pooled @ W2^T + b2.
// One block (256 threads, 8 waves) per b.
// ---------------------------------------------------------------------------
__global__ void leam_softmax_pool_out(const float* __restrict__ mIn,  // [B][S]
                                      const int*   __restrict__ gidx, // [B][S]
                                      const float* __restrict__ emb,  // [V][P]
                                      const float* __restrict__ W2,   // [K][P]
                                      const float* __restrict__ b2,   // [K]
                                      float*       __restrict__ out)  // [B][K]
{
    __shared__ float beta[Sn];
    __shared__ int   lidx[Sn];
    __shared__ float pooled[Pn];
    __shared__ float redA[8];
    __shared__ float redB[8];

    const int b    = blockIdx.x;
    const int tid  = threadIdx.x;
    const int wave = tid >> 5;
    const int lane = tid & 31;

    const float m0 = mIn[b * Sn + tid];
    const float m1 = mIn[b * Sn + 256 + tid];
    lidx[tid]       = gidx[b * Sn + tid];
    lidx[256 + tid] = gidx[b * Sn + 256 + tid];

    // block max
    float mx = fmaxf(m0, m1);
    #pragma unroll
    for (int off = 16; off > 0; off >>= 1) mx = fmaxf(mx, __shfl_down(mx, off, 32));
    if (lane == 0) redA[wave] = mx;
    __syncthreads();
    mx = redA[0];
    #pragma unroll
    for (int i = 1; i < 8; ++i) mx = fmaxf(mx, redA[i]);

    // block sum of exp
    const float e0 = __expf(m0 - mx);
    const float e1 = __expf(m1 - mx);
    float sm = e0 + e1;
    #pragma unroll
    for (int off = 16; off > 0; off >>= 1) sm += __shfl_down(sm, off, 32);
    if (lane == 0) redB[wave] = sm;
    __syncthreads();
    float tot = 0.0f;
    #pragma unroll
    for (int i = 0; i < 8; ++i) tot += redB[i];
    const float invTot = 1.0f / tot;

    beta[tid]       = e0 * invTot;
    beta[256 + tid] = e1 * invTot;
    __syncthreads();

    // pooled[p] = (1/S) * sum_s beta[s] * emb[idx[b,s]][p]   (coalesced over p)
    float f0 = 0.0f, f1 = 0.0f;
    for (int s = 0; s < Sn; ++s) {
        const float* row = emb + (size_t)lidx[s] * Pn;
        if (s + 1 < Sn)
            __builtin_prefetch(emb + (size_t)lidx[s + 1] * Pn + tid, 0, 0);
        const float w = beta[s];
        f0 += w * row[tid];                       // tid in [0,256) < 300
        if (tid < Pn - 256) f1 += w * row[256 + tid];
    }
    pooled[tid] = f0 * (1.0f / (float)Sn);
    if (tid < Pn - 256) pooled[256 + tid] = f1 * (1.0f / (float)Sn);
    __syncthreads();

    // out[b,k] = pooled . W2[k] + b2[k]; waves stride over k
    for (int k = wave; k < Kn; k += 8) {
        const float* wrow = W2 + (size_t)k * Pn;
        float acc = 0.0f;
        for (int p = lane; p < Pn; p += 32) acc += pooled[p] * wrow[p];
        #pragma unroll
        for (int off = 16; off > 0; off >>= 1) acc += __shfl_down(acc, off, 32);
        if (lane == 0) out[b * Kn + k] = acc + b2[k];
    }
}

// ---------------------------------------------------------------------------
// Launcher. Inputs: idx, emb, C, conv_w, conv_b, W2, b2  (setup_inputs order)
// Workspace layout (all 16B-aligned):
//   lhat  : Kp*Pp f16              =     40,960 B
//   xhat  : B*S*Pp f16             = 20,971,520 B
//   G     : B*Kp*S f32             =  8,388,608 B
//   mbuf  : B*S f32                =    131,072 B    (total ~28.2 MB)
// ---------------------------------------------------------------------------
extern "C" void kernel_launch(void* const* d_in, const int* in_sizes, int n_in,
                              void* d_out, int out_size, void* d_ws, size_t ws_size,
                              hipStream_t stream) {
    const int*   idx    = (const int*)  d_in[0];
    const float* emb    = (const float*)d_in[1];
    const float* C      = (const float*)d_in[2];
    const float* conv_w = (const float*)d_in[3];
    const float* conv_b = (const float*)d_in[4];
    const float* W2     = (const float*)d_in[5];
    const float* b2     = (const float*)d_in[6];
    float* out = (float*)d_out;

    char* ws = (char*)d_ws;
    _Float16* lhat = (_Float16*)(ws);
    _Float16* xhat = (_Float16*)(ws + 40960);
    float*    G    = (float*)   (ws + 40960 + 20971520);
    float*    mbuf = (float*)   (ws + 40960 + 20971520 + 8388608);

    // 1) normalize labels into padded f16 [Kp][Pp]
    leam_norm_rows<<<Kp / 8, 256, 0, stream>>>(C, nullptr, lhat, Kp, Kn);
    // 2) gather + normalize sentence embeddings into padded f16 [B*S][Pp]
    leam_norm_rows<<<(Bn * Sn) / 8, 256, 0, stream>>>(emb, idx, xhat, Bn * Sn, Bn * Sn);
    // 3) WMMA cosine-similarity GEMM -> G [B][Kp][S]
    leam_gemm_wmma<<<Bn * (Sn / 64), 256, 0, stream>>>(lhat, xhat, G);
    // 4) conv(11) + ReLU + max over labels -> m [B][S]
    leam_conv_max<<<(Bn * Sn) / 256, 256, 0, stream>>>(G, conv_w, conv_b, mbuf);
    // 5) softmax + weighted mean pool + final linear -> out [B][K]
    leam_softmax_pool_out<<<Bn, 256, 0, stream>>>(mbuf, idx, emb, W2, b2, out);
}
